// LCritic_18322330485052
// MI455X (gfx1250) — compile-verified, matrix operand order checked by reading, not run on hardware
//
#include <hip/hip_runtime.h>
#include <hip/hip_bf16.h>
#include <math.h>

typedef __attribute__((ext_vector_type(16))) __bf16       v16bf;
typedef __attribute__((ext_vector_type(8)))  float        v8f;
typedef __attribute__((ext_vector_type(4)))  unsigned int v4u;
typedef __attribute__((ext_vector_type(4)))  int          v4i;

#define AS1 __attribute__((address_space(1)))
#define AS3 __attribute__((address_space(3)))

#define B_ROWS   32768
#define HID      64
#define NCOLS    256                 // 4*HID
#define K_STATE  2048
#define K_ACT    512
#define KSTEP    32
#define NK_ST    (K_STATE / KSTEP)   // 64
#define NK_AC    (K_ACT / KSTEP)     // 16
#define NK_IN    (NK_ST + NK_AC)     // 80
#define NK_H     (HID / KSTEP)       // 2
#define NK_ALL   (NK_IN + NK_H)      // 82
#define NTILES   (NCOLS / 16)        // 16
#define TM       64                  // rows per block
#define CHUNK_DW (NTILES * 32 * 8)   // 4096 dwords = 16KB per K-step

#if defined(__gfx1250__) && __has_builtin(__builtin_amdgcn_global_load_async_to_lds_b128)
#define USE_ASYNC_LDS 1
#else
#define USE_ASYNC_LDS 0
#endif

// ---------------------------------------------------------------------------
// Pack fp32 weights W[K][256] (bf16) into WMMA B-fragment order, one
// contiguous 16KB chunk per 32-wide K-step:
//   dst dword = ((ks*NTILES + tn)*32 + lane)*8 + d
// ---------------------------------------------------------------------------
__global__ void pack_weights(const float* __restrict__ W,
                             unsigned int* __restrict__ dst, int nksteps) {
    int idx   = blockIdx.x * blockDim.x + threadIdx.x;
    int total = NTILES * nksteps * 32 * 8;
    if (idx >= total) return;
    int d    = idx & 7;
    int lane = (idx >> 3) & 31;
    int grp  = idx >> 8;             // ks*NTILES + tn
    int tn   = grp % NTILES;
    int ks   = grp / NTILES;
    int n = tn * 16 + (lane & 15);
    int g = lane >> 4;
    int k = ks * KSTEP + ((d < 4) ? (8 * g + 2 * d) : (16 + 8 * g + 2 * (d - 4)));
    float f0 = W[(size_t)k * NCOLS + n];
    float f1 = W[(size_t)(k + 1) * NCOLS + n];
    union { __bf16 b[2]; unsigned int u; } cv;
    cv.b[0] = (__bf16)f0;
    cv.b[1] = (__bf16)f1;
    dst[idx] = cv.u;
}

// Stage one 16KB packed-B chunk into LDS (256 threads, 64B each).
__device__ __forceinline__ void stage_b(const unsigned int* __restrict__ gsrc,
                                        unsigned int* lbuf, int tid) {
#if USE_ASYNC_LDS
    AS1 v4i* g = (AS1 v4i*)(gsrc + tid * 4);
    AS3 v4i* l = (AS3 v4i*)(lbuf + tid * 4);
    __builtin_amdgcn_global_load_async_to_lds_b128(g, l, 0,     0);
    __builtin_amdgcn_global_load_async_to_lds_b128(g, l, 4096,  0);
    __builtin_amdgcn_global_load_async_to_lds_b128(g, l, 8192,  0);
    __builtin_amdgcn_global_load_async_to_lds_b128(g, l, 12288, 0);
#else
    const v4u* s = (const v4u*)gsrc + tid;
    v4u*       p = (v4u*)lbuf + tid;
#pragma unroll
    for (int j = 0; j < 4; ++j) p[j * 256] = s[j * 256];
#endif
}

__device__ __forceinline__ void wait_async_zero() {
#if USE_ASYNC_LDS
#if __has_builtin(__builtin_amdgcn_s_wait_asynccnt)
    __builtin_amdgcn_s_wait_asynccnt(0);
#else
    asm volatile("s_wait_asynccnt 0x0" ::: "memory");
#endif
#endif
}

// Load 32 consecutive-K fp32 of one row, convert to a CDNA5 16-bit A fragment.
__device__ __forceinline__ v16bf load_a_frag(const float* __restrict__ rowp,
                                             int kb, int g) {
    const float4* p0 = (const float4*)(rowp + kb + 8 * g);
    const float4* p1 = (const float4*)(rowp + kb + 16 + 8 * g);
    float4 a0 = p0[0], a1 = p0[1];
    float4 a2 = p1[0], a3 = p1[1];
    v16bf a;
    a[0]  = (__bf16)a0.x; a[1]  = (__bf16)a0.y; a[2]  = (__bf16)a0.z; a[3]  = (__bf16)a0.w;
    a[4]  = (__bf16)a1.x; a[5]  = (__bf16)a1.y; a[6]  = (__bf16)a1.z; a[7]  = (__bf16)a1.w;
    a[8]  = (__bf16)a2.x; a[9]  = (__bf16)a2.y; a[10] = (__bf16)a2.z; a[11] = (__bf16)a2.w;
    a[12] = (__bf16)a3.x; a[13] = (__bf16)a3.y; a[14] = (__bf16)a3.z; a[15] = (__bf16)a3.w;
    return a;
}

__device__ __forceinline__ float fast_sigmoid(float x) {
    return 1.0f / (1.0f + __expf(-x));
}
__device__ __forceinline__ float fast_tanh(float x) {
    float e = __expf(2.0f * x);
    return 1.0f - 2.0f / (e + 1.0f);
}

__global__ __launch_bounds__(256)
void lstm_critic_kernel(const float* __restrict__ state,
                        const float* __restrict__ action,
                        const float* __restrict__ hidden,
                        const float* __restrict__ cell,
                        const float* __restrict__ bi,
                        const float* __restrict__ bh,
                        const float* __restrict__ Wo,
                        const float* __restrict__ bo,
                        const unsigned int* __restrict__ WP,
                        float* __restrict__ out) {
    __shared__ __align__(32) unsigned int bstage[2][CHUNK_DW]; // 2 x 16KB
    __shared__ float gbuf[TM][NCOLS + 4];                      // gates, then h_t

    const int tid  = threadIdx.x;
    const int lane = tid & 31;
    const int wave = tid >> 5;
    const int mh   = wave >> 2;   // 0..1 : which 32-row half
    const int q    = wave & 3;    // 0..3 : which 64-col gate chunk
    const int g    = lane >> 4;
    const int nl   = lane & 15;

    const int    row0 = blockIdx.x * TM;
    const size_t m0   = (size_t)(row0 + mh * 32 + nl);
    const size_t m1   = m0 + 16;

    const float* s0 = state  + m0 * (size_t)K_STATE;
    const float* s1 = state  + m1 * (size_t)K_STATE;
    const float* a0 = action + m0 * (size_t)K_ACT;
    const float* a1 = action + m1 * (size_t)K_ACT;
    const float* h0 = hidden + m0 * (size_t)HID;
    const float* h1 = hidden + m1 * (size_t)HID;

    v8f accA[4] = {};   // rows m0 fragment
    v8f accB[4] = {};   // rows m1 fragment

    const int boff = ((q * 4) * 32 + lane) * 8;   // this wave's first B frag

    // One GEMM K-step: preload all 4 B fragments, then 8 back-to-back WMMAs.
    auto gemm_step = [&](const float* r0, const float* r1, int kb,
                         const unsigned int* bc) {
        __builtin_prefetch(r0 + kb + 4 * KSTEP + 8 * g, 0, 1);
        v16bf fa0 = load_a_frag(r0, kb, g);
        v16bf fa1 = load_a_frag(r1, kb, g);
        v16bf fb[4];
#pragma unroll
        for (int t = 0; t < 4; ++t)
            fb[t] = *(const v16bf*)(bc + boff + t * 32 * 8);
#pragma unroll
        for (int t = 0; t < 4; ++t) {
            accA[t] = __builtin_amdgcn_wmma_f32_16x16x32_bf16(
                false, fa0, false, fb[t], (short)0, accA[t], false, false);
            accB[t] = __builtin_amdgcn_wmma_f32_16x16x32_bf16(
                false, fa1, false, fb[t], (short)0, accB[t], false, false);
        }
    };

    // ---- staging pipeline: issue K-step 0, then always stay 1 step ahead ----
    stage_b(WP, bstage[0], tid);
    const unsigned int* wp_next = WP + CHUNK_DW;
    int buf = 0;

#pragma unroll 2
    for (int ks = 0; ks < NK_ST; ++ks) {          // x(state) @ Wi
        wait_async_zero();
        __syncthreads();
        stage_b(wp_next, bstage[buf ^ 1], tid);
        wp_next += CHUNK_DW;
        gemm_step(s0, s1, ks * KSTEP, bstage[buf]);
        buf ^= 1;
    }
#pragma unroll 2
    for (int ks = 0; ks < NK_AC; ++ks) {          // x(action) @ Wi
        wait_async_zero();
        __syncthreads();
        stage_b(wp_next, bstage[buf ^ 1], tid);
        wp_next += CHUNK_DW;
        gemm_step(a0, a1, ks * KSTEP, bstage[buf]);
        buf ^= 1;
    }
#pragma unroll
    for (int ks = 0; ks < NK_H; ++ks) {           // h @ Wh
        wait_async_zero();
        __syncthreads();
        if (ks + 1 < NK_H) {
            stage_b(wp_next, bstage[buf ^ 1], tid);
            wp_next += CHUNK_DW;
        }
        gemm_step(h0, h1, ks * KSTEP, bstage[buf]);
        buf ^= 1;
    }

    // ---- add biases, spill gates to LDS (C layout: lane = N, vgpr r = M) ----
#pragma unroll
    for (int t = 0; t < 4; ++t) {
        int   ncol = (q * 4 + t) * 16 + nl;
        float bias = bi[ncol] + bh[ncol];
#pragma unroll
        for (int r = 0; r < 8; ++r) {
            gbuf[mh * 32 + g * 8 + r][ncol]      = accA[t][r] + bias;
            gbuf[mh * 32 + 16 + g * 8 + r][ncol] = accB[t][r] + bias;
        }
    }
    __syncthreads();

    // ---- LSTM elementwise; h_t written back into gbuf cols [0,64) ----
    for (int e = tid; e < TM * HID; e += 256) {
        int    mr   = e >> 6;
        int    j    = e & 63;
        size_t grow = (size_t)row0 + mr;
        float iv = fast_sigmoid(gbuf[mr][j]);
        float fv = fast_sigmoid(gbuf[mr][HID + j]);
        float gv = fast_tanh(gbuf[mr][2 * HID + j]);
        float ov = fast_sigmoid(gbuf[mr][3 * HID + j]);
        float cp = cell[grow * HID + j];
        float ct = fv * cp + iv * gv;
        float ht = ov * fast_tanh(ct);
        out[(size_t)B_ROWS + grow * HID + j]                        = ht;  // h_t
        out[(size_t)B_ROWS + (size_t)B_ROWS * HID + grow * HID + j] = ct;  // c_t
        gbuf[mr][j] = ht;
    }
    __syncthreads();

    // ---- out = tanh(h_t @ Wo + bo) ----
    if (tid < TM) {
        float a = bo[0];
#pragma unroll 8
        for (int j = 0; j < HID; ++j) a += gbuf[tid][j] * Wo[j];
        out[row0 + tid] = fast_tanh(a);
    }
}

extern "C" void kernel_launch(void* const* d_in, const int* in_sizes, int n_in,
                              void* d_out, int out_size, void* d_ws, size_t ws_size,
                              hipStream_t stream) {
    const float* state  = (const float*)d_in[0];
    const float* action = (const float*)d_in[1];
    const float* hidden = (const float*)d_in[2];
    const float* cell   = (const float*)d_in[3];
    const float* Wi     = (const float*)d_in[4];
    const float* bi     = (const float*)d_in[5];
    const float* Wh     = (const float*)d_in[6];
    const float* bh     = (const float*)d_in[7];
    const float* Wo     = (const float*)d_in[8];
    const float* bo     = (const float*)d_in[9];
    float* out = (float*)d_out;

    // One packed array: K-steps 0..79 from Wi, 80..81 from Wh  (~1.35 MB)
    unsigned int* WP = (unsigned int*)d_ws;

    {
        int total = NTILES * NK_IN * 32 * 8;
        pack_weights<<<(total + 255) / 256, 256, 0, stream>>>(Wi, WP, NK_IN);
    }
    {
        int total = NTILES * NK_H * 32 * 8;
        pack_weights<<<(total + 255) / 256, 256, 0, stream>>>(
            Wh, WP + (size_t)NK_IN * CHUNK_DW, NK_H);
    }
    lstm_critic_kernel<<<B_ROWS / TM, 256, 0, stream>>>(
        state, action, hidden, cell, bi, bh, Wo, bo, WP, out);
}
